// spatial_embedding_28406913695895
// MI455X (gfx1250) — matrix operand order repaired and checked by reading, not compiled, not run on hardware
//
#include <hip/hip_runtime.h>

#define J  25
#define JP 32
#define H  256

typedef _Float16 h8  __attribute__((ext_vector_type(8)));
typedef _Float16 h16 __attribute__((ext_vector_type(16)));
typedef float    f8  __attribute__((ext_vector_type(8)));
typedef float    f4v __attribute__((ext_vector_type(4)));

__global__ __launch_bounds__(256)
void gat_frame_kernel(const float* __restrict__ x,
                      const int*   __restrict__ adj,
                      const float* __restrict__ W,
                      const float* __restrict__ a,
                      float*       __restrict__ out,
                      int n_frames)
{
    __shared__ __align__(64) float    sx[J * 3];
    __shared__ __align__(64) _Float16 sWhT[H][JP];   // [col h][k=j], K-padded w/ zeros
    __shared__ __align__(64) _Float16 sAtt[JP][JP];  // [row i][k=j], padded w/ zeros
    __shared__ __align__(64) float    sS[JP];
    __shared__ __align__(64) float    sRed[8][J];
    __shared__ __align__(64) float    sOut[JP * H];  // D staging, rows 25..31 = pad

    const int f = blockIdx.x;
    if (f >= n_frames) return;

    const int tid  = threadIdx.x;    // 0..255, == column h
    const int lane = tid & 31;
    const int wave = tid >> 5;

    // ---- load frame x (25*3 floats), W column h, a[h] ----
    if (tid < J * 3) sx[tid] = x[(size_t)f * (J * 3) + tid];
    const int   h  = tid;
    const float w0 = W[0 * H + h];
    const float w1 = W[1 * H + h];
    const float w2 = W[2 * H + h];
    const float ah = a[h];           // reference uses a[:H] only
    __syncthreads();

    // ---- Wh[j][h] in f32; stash f16 transposed for WMMA B; partial s[j] ----
    float ps[J];
#pragma unroll
    for (int j = 0; j < J; ++j) {
        const float wh = sx[3 * j] * w0 + sx[3 * j + 1] * w1 + sx[3 * j + 2] * w2;
        sWhT[h][j] = (_Float16)wh;
        ps[j] = wh * ah;
    }
#pragma unroll
    for (int j = J; j < JP; ++j) sWhT[h][j] = (_Float16)0.f;   // K pad

    // ---- s[j] = sum_h Wh[j][h]*a[h]: wave shuffle tree + cross-wave LDS ----
#pragma unroll
    for (int j = 0; j < J; ++j) {
        float v = ps[j];
        for (int o = 16; o > 0; o >>= 1) v += __shfl_down(v, o, 32);
        if (lane == 0) sRed[wave][j] = v;
    }
    __syncthreads();
    if (tid < J) {
        float s = 0.f;
#pragma unroll
        for (int w = 0; w < 8; ++w) s += sRed[w][tid];
        sS[tid] = s;
    }
    __syncthreads();

    // ---- masked relu + row softmax -> att (f16, zero padded to 32x32) ----
    if (tid < J) {
        const int   i  = tid;
        const float si = sS[i];
        float e[J];
        float m = 0.f;
#pragma unroll
        for (int j = 0; j < J; ++j) {
            const float v = (adj[i * J + j] == 1) ? fmaxf(si + sS[j], 0.f) : 0.f;
            e[j] = v;
            m = fmaxf(m, v);
        }
        float sum = 0.f;
#pragma unroll
        for (int j = 0; j < J; ++j) { e[j] = __expf(e[j] - m); sum += e[j]; }
        const float inv = 1.f / sum;
#pragma unroll
        for (int j = 0; j < J; ++j) sAtt[i][j] = (_Float16)(e[j] * inv);
#pragma unroll
        for (int j = J; j < JP; ++j) sAtt[i][j] = (_Float16)0.f;
    } else if (tid < JP) {
#pragma unroll
        for (int j = 0; j < JP; ++j) sAtt[tid][j] = (_Float16)0.f;  // M pad rows
    }
    __syncthreads();

    // ---- out = att(32x32) @ Wh(32x256) via v_wmma_f32_16x16x32_f16 ----
    // 2 M-tiles x 16 N-tiles = 32 WMMAs; 4 per wave. K=32 in one instruction.
    const int mtile = wave >> 2;         // waves 0-3 -> rows 0..15, waves 4-7 -> 16..31
    const int nbase = (wave & 3) * 4;    // 4 consecutive N-tiles per wave
    const int hl    = lane & 15;
    const int hi    = lane >> 4;

    // A fragment: row = mtile*16 + hl; halves 0..7 = K k0..k0+7, 8..15 = K k0+16..k0+23
    const int arow = mtile * 16 + hl;
    const int k0   = hi ? 8 : 0;
    h16 afrag;
    {
        h8* ap = (h8*)&afrag;
        ap[0] = *(const h8*)&sAtt[arow][k0];
        ap[1] = *(const h8*)&sAtt[arow][k0 + 16];
    }

#pragma unroll
    for (int nt = 0; nt < 4; ++nt) {
        const int ntile = nbase + nt;
        const int col   = ntile * 16 + hl;
        const int k0b   = hi ? 8 : 0;    // lanes 0-15: K 0..7,16..23, lanes 16-31: K 8..15,24..31
        h16 bfrag;
        h8* bp = (h8*)&bfrag;
        bp[0] = *(const h8*)&sWhT[col][k0b];
        bp[1] = *(const h8*)&sWhT[col][k0b + 16];

        f8 c = {};
        c = __builtin_amdgcn_wmma_f32_16x16x32_f16(false, afrag, false, bfrag,
                                                   (short)0, c, false, false);

        // D layout: lane -> column `col`; VGPR r -> row r + 8*hi (+ mtile*16).
        // Stage to LDS (rows >= 25 land in pad rows, never copied out) -> no guards.
        const int rbase = mtile * 16 + 8 * hi;
#pragma unroll
        for (int r = 0; r < 8; ++r) {
            sOut[(rbase + r) * H + col] = c[r];
        }
    }
    __syncthreads();

    // ---- branch-free, fully coalesced 128-bit non-temporal stores ----
    const f4v* src = (const f4v*)sOut;
    f4v*       dst = (f4v*)(out + (size_t)f * (J * H));
    for (int i = tid; i < (J * H) / 4; i += 256) {        // 1600 float4 per frame
        __builtin_nontemporal_store(src[i], &dst[i]);
    }
}

extern "C" void kernel_launch(void* const* d_in, const int* in_sizes, int n_in,
                              void* d_out, int out_size, void* d_ws, size_t ws_size,
                              hipStream_t stream)
{
    const float* x   = (const float*)d_in[0];   // (B,T,75)
    const int*   adj = (const int*)  d_in[1];   // (25,25)
    const float* W   = (const float*)d_in[2];   // (3,256)
    const float* a   = (const float*)d_in[3];   // (512,1)
    float*       out = (float*)d_out;           // (B,T,25,256)

    const int n_frames = in_sizes[0] / (J * 3); // B*T = 9600
    gat_frame_kernel<<<n_frames, 256, 0, stream>>>(x, adj, W, a, out, n_frames);
}
